// LSTM_23725399343359
// MI455X (gfx1250) — matrix operand order repaired and checked by reading, not compiled, not run on hardware
//
#include <hip/hip_runtime.h>

// ---------------------------------------------------------------------------
// 2-layer LSTM LM on MI455X (gfx1250, wave32, WMMA, TDM, async-LDS).
// bf16 operands / f32 accumulate via v_wmma_f32_16x16x32_bf16.
// All WMMA operands live in fragment-linear packed buffers (1 tile = 32
// lanes * 16 bf16 contiguous = 1KB), so a fragment load is one aligned
// 32-byte v16bf read per lane (2x b128).
// ---------------------------------------------------------------------------

typedef __attribute__((ext_vector_type(16))) __bf16   v16bf;
typedef __attribute__((ext_vector_type(8)))  float    v8f;
typedef __attribute__((ext_vector_type(4)))  unsigned v4u;
typedef __attribute__((ext_vector_type(8)))  int      v8i;
typedef __attribute__((ext_vector_type(4)))  int      v4i;

#define DI __device__ __forceinline__

static constexpr int TT = 512;   // time steps
static constexpr int BB = 64;    // batch
static constexpr int HH = 1024;  // hidden
static constexpr int EE = 512;   // embedding
static constexpr int NG = 4 * HH;   // 4096 gate columns
static constexpr int MM = TT * BB;  // 32768 GEMM rows

// LDS layout for the recurrence kernel (dynamic shared)
static constexpr unsigned LDS_STAGE_B = 131072;              // h A-fragments
static constexpr unsigned LDS_LH_B    = 64 * 65 * 4;         // 16640
static constexpr unsigned LDS_PK_B    = 8 * 512 * 2;         // 8192
static constexpr unsigned LDS_TOTAL_B = LDS_STAGE_B + LDS_LH_B + LDS_PK_B;

#if defined(__has_builtin)
#if __has_builtin(__builtin_amdgcn_tensor_load_to_lds)
#define HAVE_TDM 1
#endif
#endif
#ifndef HAVE_TDM
#define HAVE_TDM 0
#endif

DI v8f wmma_bf16(v16bf a, v16bf b, v8f c) {
  return __builtin_amdgcn_wmma_f32_16x16x32_bf16(false, a, false, b, (short)0, c,
                                                 false, false);
}

// A-matrix 16x32 bf16 lane map (ISA 7.12.2): lane L (hf=L>>4, m=L&15),
// element e -> k offset within the 32-wide tile.
DI int a_koff(int hf, int e) {
  return (e < 8) ? (hf * 8 + e) : (16 + hf * 8 + (e - 8));
}

#if HAVE_TDM
// 1-row TDM copy: global -> LDS, `bytes` must be a multiple of 8, <= 512KB.
// D# packing per CDNA5 ISA ch.8: group0 {count=1, lds_addr, global_addr,
// type=2}; group1 {data_size=8B, tensor_dim0=units, tensor_dim1=1,
// tile_dim0=units, tile_dim1=1, dim0_stride=units}.  Groups 2/3 unused
// (<=2D tensor); this toolchain's builtin takes 6 args (extra int32x8).
DI void tdm_load_1d(const void* gsrc, unsigned lds_off, unsigned bytes) {
  unsigned units = bytes >> 3;  // 8-byte elements
  unsigned long long ga = (unsigned long long)gsrc;
  v4u g0;
  g0[0] = 1u;                                    // count=1, user mode
  g0[1] = lds_off;                               // lds_addr
  g0[2] = (unsigned)ga;                          // global_addr[31:0]
  g0[3] = (unsigned)(ga >> 32) | (2u << 30);     // global_addr[56:32] | type=2
  v8i g1;
  g1[0] = (3 << 16);                             // data_size=3 (8B)
  g1[1] = (int)((units & 0xffffu) << 16);        // tensor_dim0[15:0] @ 63:48
  g1[2] = (int)((units >> 16) | (1u << 16));     // dim0[31:16] | tensor_dim1=1
  g1[3] = (int)((units & 0xffffu) << 16);        // tile_dim0 @ 127:112
  g1[4] = 1;                                     // tile_dim1=1, tile_dim2=0
  g1[5] = (int)units;                            // tensor_dim0_stride[31:0]
  g1[6] = 0;
  g1[7] = 0;
  v4i z4 = {0, 0, 0, 0};
  v8i z8 = {0, 0, 0, 0, 0, 0, 0, 0};
  __builtin_amdgcn_tensor_load_to_lds(g0, g1, z4, z4, z8, 0);
}
#endif

// ---------------------------------------------------------------------------
// Pack W [N x K] row-major f32 into B-fragments for gates = x @ W^T.
// ---------------------------------------------------------------------------
__global__ void k_pack_B(const float* __restrict__ W, __bf16* __restrict__ out,
                         int N, int K) {
  int gid = blockIdx.x * 256 + threadIdx.x;
  int KT = K >> 5;
  int total = (N >> 4) * KT * 32;
  if (gid >= total) return;
  int lane = gid & 31;
  int tile = gid >> 5;
  int n  = (tile / KT) * 16 + (lane & 15);
  int kb = (tile % KT) * 32;
  const float* src = W + (size_t)n * K + kb;
  __bf16* dst = out + (size_t)tile * 512 + lane * 16;
#pragma unroll
  for (int e = 0; e < 16; ++e) dst[e] = (__bf16)src[a_koff(lane >> 4, e)];
}

// Embedding gather + pack into A-fragments: row m = t*B+b of x = emb[tok[m]].
__global__ void k_embed_pack_A(const int* __restrict__ tok,
                               const float* __restrict__ emb,
                               __bf16* __restrict__ out, int M, int K) {
  int gid = blockIdx.x * 256 + threadIdx.x;
  int KT = K >> 5;
  int total = (M >> 4) * KT * 32;
  if (gid >= total) return;
  int lane = gid & 31, tile = gid >> 5;
  int mt = tile / KT, kt = tile % KT;
  int m = mt * 16 + (lane & 15);
  int hf = lane >> 4;
  const float* row = emb + (size_t)tok[m] * K + kt * 32;
  __bf16* dst = out + (size_t)tile * 512 + lane * 16;
#pragma unroll
  for (int e = 0; e < 16; ++e) dst[e] = (__bf16)row[a_koff(hf, e)];
}

// Pack a plain f32 [M x K] row-major matrix into A-fragments (y0 -> A1).
__global__ void k_pack_A(const float* __restrict__ X, __bf16* __restrict__ out,
                         int M, int K) {
  int gid = blockIdx.x * 256 + threadIdx.x;
  int KT = K >> 5;
  int total = (M >> 4) * KT * 32;
  if (gid >= total) return;
  int lane = gid & 31, tile = gid >> 5;
  int mt = tile / KT, kt = tile % KT;
  int m = mt * 16 + (lane & 15);
  int hf = lane >> 4;
  const float* row = X + (size_t)m * K + kt * 32;
  __bf16* dst = out + (size_t)tile * 512 + lane * 16;
#pragma unroll
  for (int e = 0; e < 16; ++e) dst[e] = (__bf16)row[a_koff(hf, e)];
}

// ---------------------------------------------------------------------------
// GEMM: D[M x N] = A(frags) x B(frags).  256 thr = 8 waves, 4x2 layout;
// wave tile 32x64 (2x4 C-frags, 8 WMMAs / K-step), WG tile 128x128.
// KT is a compile-time constant so all fragment addresses fold into
// immediate offsets of the b128 loads (no per-iteration 64-bit adds).
// ---------------------------------------------------------------------------
template <int KT>
__global__ void __launch_bounds__(256) k_gemm(const __bf16* __restrict__ Af,
                                              const __bf16* __restrict__ Bf,
                                              float* __restrict__ D, int N) {
  int lane = threadIdx.x & 31, wave = threadIdx.x >> 5;
  int wm = wave & 3, wn = wave >> 2;
  int mt0 = blockIdx.x * 8 + wm * 2;
  int nt0 = blockIdx.y * 8 + wn * 4;
  const __bf16* ap = Af + (size_t)mt0 * KT * 512 + lane * 16;
  const __bf16* bp = Bf + (size_t)nt0 * KT * 512 + lane * 16;
  v8f acc[2][4] = {};
#pragma unroll 4
  for (int kt = 0; kt < KT; ++kt) {
    v16bf a0 = *(const v16bf*)(ap + kt * 512);
    v16bf a1 = *(const v16bf*)(ap + (KT + kt) * 512);
    v16bf b0 = *(const v16bf*)(bp + kt * 512);
    v16bf b1 = *(const v16bf*)(bp + (KT + kt) * 512);
    v16bf b2 = *(const v16bf*)(bp + (2 * KT + kt) * 512);
    v16bf b3 = *(const v16bf*)(bp + (3 * KT + kt) * 512);
    __builtin_prefetch(ap + (kt + 1) * 512, 0, 3);
    __builtin_prefetch(bp + (kt + 1) * 512, 0, 3);
    acc[0][0] = wmma_bf16(a0, b0, acc[0][0]);
    acc[0][1] = wmma_bf16(a0, b1, acc[0][1]);
    acc[0][2] = wmma_bf16(a0, b2, acc[0][2]);
    acc[0][3] = wmma_bf16(a0, b3, acc[0][3]);
    acc[1][0] = wmma_bf16(a1, b0, acc[1][0]);
    acc[1][1] = wmma_bf16(a1, b1, acc[1][1]);
    acc[1][2] = wmma_bf16(a1, b2, acc[1][2]);
    acc[1][3] = wmma_bf16(a1, b3, acc[1][3]);
  }
  int rr = (lane >> 4) * 8, col = lane & 15;
#pragma unroll
  for (int i = 0; i < 2; ++i)
#pragma unroll
    for (int jn = 0; jn < 4; ++jn)
#pragma unroll
      for (int v = 0; v < 8; ++v)
        D[(size_t)((mt0 + i) * 16 + rr + v) * N + (nt0 + jn) * 16 + col] =
            acc[i][jn][v];
}

// ---------------------------------------------------------------------------
// Device-wide step barrier for the 16 persistent recurrence workgroups.
// ---------------------------------------------------------------------------
DI void gsync(unsigned* cnt, unsigned phase, int tid) {
  __syncthreads();
  if (tid == 0) {
    atomicAdd(cnt, 1u);
    unsigned target = phase * 16u;
    while (((volatile unsigned*)cnt)[0] < target) __builtin_amdgcn_s_sleep(1);
  }
  __syncthreads();
  __builtin_amdgcn_s_cluster_barrier();  // NOP when not cluster-launched
}

// ---------------------------------------------------------------------------
// Persistent LSTM recurrence.  16 WGs; WG j owns hidden units [j*64,j*64+64).
// Cell state in C-fragments (registers) for all T steps.  Per step:
//   1) TDM (tensor_load_to_lds) stages the 128KB packed h buffer into LDS
//   2) gates = x_proj[t] + h @ W_hh^T : WMMA fed by ds_load from LDS
//   3) activations; h re-packed in LDS, pushed with
//      global_store_async_from_lds_b128 (scope DEV) + s_wait_asynccnt
//   4) device-wide arrive barrier (+ s_cluster_barrier)
// ---------------------------------------------------------------------------
__global__ void __launch_bounds__(256) k_rec(
    const float* __restrict__ xproj,  // [T*64 x 4096]
    const __bf16* __restrict__ Whh,   // packed B frags, KT=32
    const float* __restrict__ h0,     // [64 x 1024]
    const float* __restrict__ c0,     // [64 x 1024]
    __bf16* __restrict__ hfrag,       // double buffer: 2 * 65536 bf16
    float* __restrict__ yout,         // [T*64 x 1024] or nullptr
    float* __restrict__ pooled,       // [64 x 1024] sum over t, or nullptr
    float* __restrict__ hn, float* __restrict__ cn,
    unsigned* __restrict__ counter, int T) {
  constexpr int FRAG = 512;
  constexpr int HB = 4 * 32 * FRAG;  // elements per h buffer
  extern __shared__ char smem[];
  __bf16* stage = (__bf16*)smem;                          // 128KB h frags
  float*  lh    = (float*)(smem + LDS_STAGE_B);           // h staging (pad 65)
  __bf16* pk    = (__bf16*)(smem + LDS_STAGE_B + LDS_LH_B);  // pack staging

  int tid = threadIdx.x, lane = tid & 31, wave = tid >> 5;
  int j = blockIdx.x;  // 0..15
  int wm = wave & 3;   // batch tile
  int up = wave >> 2;  // unit-tile pair selector (0..1)
  int hf = lane >> 4, rr = hf * 8, col = lane & 15;

  v8f cst[2], ps[2];
#pragma unroll
  for (int p = 0; p < 2; ++p) {
    int u0 = j * 64 + (up * 2 + p) * 16;
    v8f v;
#pragma unroll
    for (int e = 0; e < 8; ++e)
      v[e] = c0[(size_t)(wm * 16 + rr + e) * HH + u0 + col];
    cst[p] = v;
    v8f z = {};
    ps[p] = z;
  }

  // Hoisted per-(p,g) W_hh B-fragment base pointers: inner K-loop becomes
  // pure constant-offset loads.
  const __bf16* bptr[2][4];
#pragma unroll
  for (int p = 0; p < 2; ++p)
#pragma unroll
    for (int g = 0; g < 4; ++g) {
      int nt = g * 64 + j * 4 + up * 2 + p;
      bptr[p][g] = Whh + (size_t)nt * 32 * FRAG + lane * 16;
    }
  const __bf16* aptr = stage + (size_t)wm * 32 * FRAG + lane * 16;

  // Pack h0 into buffer 0 (this WG's kt slices 2j, 2j+1).
  {
    int mt = wave & 3, kl = wave >> 2;
    int m = mt * 16 + (lane & 15);
    __bf16* dst = hfrag + (size_t)(mt * 32 + 2 * j + kl) * FRAG + lane * 16;
#pragma unroll
    for (int e = 0; e < 16; ++e) {
      int klocal = kl * 32 + a_koff(hf, e);
      dst[e] = (__bf16)h0[(size_t)m * HH + j * 64 + klocal];
    }
  }
  __threadfence();
  unsigned phase = 0;
  gsync(counter, ++phase, tid);

  unsigned stage_lds = (unsigned)(size_t)(void*)stage;

  for (int t = 0; t < T; ++t) {
    const __bf16* rbuf = hfrag + (size_t)(t & 1) * HB;
    __bf16* wbuf = hfrag + (size_t)((t & 1) ^ 1) * HB;

    // ---- stage this step's h fragments into LDS ----
#if HAVE_TDM
    if (wave == 0) {
      tdm_load_1d(rbuf, stage_lds, LDS_STAGE_B);
      __builtin_amdgcn_s_wait_tensorcnt(0);
    }
#else
    {
      const char* gb = (const char*)rbuf;
#pragma unroll 4
      for (int c = tid; c < (int)(LDS_STAGE_B / 16); c += 256) {
        unsigned la = stage_lds + (unsigned)c * 16u;
        const char* ga = gb + (size_t)c * 16;
        asm volatile("global_load_async_to_lds_b128 %0, %1, off" ::"v"(la),
                         "v"(ga)
                     : "memory");
      }
      asm volatile("s_wait_asynccnt 0" ::: "memory");
    }
#endif
    __syncthreads();

    // ---- C-init = x_proj[t] slice ----
    v8f acc[2][4];
    const float* xp = xproj + (size_t)t * 64 * NG;
#pragma unroll
    for (int p = 0; p < 2; ++p)
#pragma unroll
      for (int g = 0; g < 4; ++g) {
        int nc = g * HH + j * 64 + (up * 2 + p) * 16 + col;
        v8f v;
#pragma unroll
        for (int e = 0; e < 8; ++e)
          v[e] = xp[(size_t)(wm * 16 + rr + e) * NG + nc];
        acc[p][g] = v;
      }

    // ---- gates += h @ W_hh^T : A from LDS (ds_load), B from L2 ----
#pragma unroll 4
    for (int kt = 0; kt < 32; ++kt) {
      v16bf a = *(const v16bf*)(aptr + kt * FRAG);
#pragma unroll
      for (int p = 0; p < 2; ++p)
#pragma unroll
        for (int g = 0; g < 4; ++g)
          acc[p][g] = wmma_bf16(a, *(const v16bf*)(bptr[p][g] + kt * FRAG),
                                acc[p][g]);
    }

    // ---- activations + cell update ----
#pragma unroll
    for (int p = 0; p < 2; ++p) {
      v8f hvv;
#pragma unroll
      for (int e = 0; e < 8; ++e) {
        float iv = 1.f / (1.f + __expf(-acc[p][0][e]));
        float fv = 1.f / (1.f + __expf(-acc[p][1][e]));
        float gv = tanhf(acc[p][2][e]);
        float ov = 1.f / (1.f + __expf(-acc[p][3][e]));
        float cv = fv * cst[p][e] + iv * gv;
        cst[p][e] = cv;
        float hv = ov * tanhf(cv);
        hvv[e] = hv;
        ps[p][e] += hv;
        int b = wm * 16 + rr + e;
        int ul = (up * 2 + p) * 16 + col;
        lh[b * 65 + ul] = hv;
        if (yout) yout[(size_t)(t * 64 + b) * HH + j * 64 + ul] = hv;
      }
      if (t == T - 1) {
#pragma unroll
        for (int e = 0; e < 8; ++e) {
          int b = wm * 16 + rr + e;
          size_t o = (size_t)b * HH + j * 64 + (up * 2 + p) * 16 + col;
          hn[o] = hvv[e];
          cn[o] = cst[p][e];
        }
      }
    }
    __syncthreads();

    // ---- re-pack h -> next-step A-frags; push via async store-from-LDS ----
    {
      int mt = wave & 3, kl = wave >> 2;
      int m = mt * 16 + (lane & 15);
      __bf16* dstl = pk + (size_t)wave * FRAG + lane * 16;
#pragma unroll
      for (int e = 0; e < 16; ++e) {
        int klocal = kl * 32 + a_koff(hf, e);
        dstl[e] = (__bf16)lh[m * 65 + klocal];
      }
      asm volatile("s_wait_dscnt 0" ::: "memory");  // LDS writes -> async engine
      unsigned la = (unsigned)(size_t)(void*)dstl;
      __bf16* gdst = wbuf + (size_t)(mt * 32 + 2 * j + kl) * FRAG + lane * 16;
      asm volatile(
          "global_store_async_from_lds_b128 %0, %1, off scope:SCOPE_DEV\n\t"
          "global_store_async_from_lds_b128 %0, %1, off offset:16 scope:SCOPE_DEV" ::
              "v"(gdst),
          "v"(la)
          : "memory");
      asm volatile("s_wait_asynccnt 0" ::: "memory");
    }
    __threadfence();
    gsync(counter, ++phase, tid);
  }

  if (pooled) {
#pragma unroll
    for (int p = 0; p < 2; ++p)
#pragma unroll
      for (int e = 0; e < 8; ++e)
        pooled[(size_t)(wm * 16 + rr + e) * HH + j * 64 + (up * 2 + p) * 16 +
               col] = ps[p][e];
  }
}

// decoded[b] = (pooled_sum[b,:] . W)/T + bias
__global__ void k_decode(const float* __restrict__ pooled,
                         const float* __restrict__ W,
                         const float* __restrict__ bias,
                         float* __restrict__ outp, int T) {
  __shared__ float red[256];
  int b = blockIdx.x;
  float s = 0.f;
  for (int u = threadIdx.x; u < HH; u += 256)
    s += pooled[(size_t)b * HH + u] * W[u];
  red[threadIdx.x] = s;
  __syncthreads();
  for (int k = 128; k > 0; k >>= 1) {
    if (threadIdx.x < k) red[threadIdx.x] += red[threadIdx.x + k];
    __syncthreads();
  }
  if (threadIdx.x == 0) outp[b] = red[0] / (float)T + bias[0];
}

__global__ void k_init(unsigned* c) {
  if (threadIdx.x < 8) c[threadIdx.x] = 0u;
}

// ---------------------------------------------------------------------------
extern "C" void kernel_launch(void* const* d_in, const int* in_sizes, int n_in,
                              void* d_out, int out_size, void* d_ws,
                              size_t ws_size, hipStream_t stream) {
  const int*   tok  = (const int*)d_in[0];
  const float* h0   = (const float*)d_in[1];
  const float* c0   = (const float*)d_in[2];
  const float* emb  = (const float*)d_in[3];
  const float* Wih0 = (const float*)d_in[4];
  const float* Whh0 = (const float*)d_in[5];
  const float* Wih1 = (const float*)d_in[6];
  const float* Whh1 = (const float*)d_in[7];
  const float* decW = (const float*)d_in[8];
  const float* decB = (const float*)d_in[9];
  float* out = (float*)d_out;
  float* o_decoded = out;
  float* o_hn = out + 64;
  float* o_cn = out + 64 + 2 * BB * HH;

  char* w = (char*)d_ws;
  size_t off = 0;
  auto carve = [&](size_t bytes) {
    char* p = w + off;
    off += (bytes + 255) & ~(size_t)255;
    return p;
  };
  unsigned* ctr    = (unsigned*)carve(256);
  __bf16*   hfr    = (__bf16*)carve(2 * 4 * 32 * 512 * sizeof(__bf16));
  float*    pooled = (float*)carve((size_t)BB * HH * sizeof(float));
  __bf16*   Bih0   = (__bf16*)carve((size_t)NG * EE * sizeof(__bf16));
  __bf16*   Bhh0   = (__bf16*)carve((size_t)NG * HH * sizeof(__bf16));
  __bf16*   Bih1   = (__bf16*)carve((size_t)NG * HH * sizeof(__bf16));
  __bf16*   Bhh1   = (__bf16*)carve((size_t)NG * HH * sizeof(__bf16));
  __bf16*   A0     = (__bf16*)carve((size_t)MM * EE * sizeof(__bf16));
  __bf16*   A1     = (__bf16*)carve((size_t)MM * HH * sizeof(__bf16));
  float*    y0     = (float*)carve((size_t)MM * HH * sizeof(float));
  float*    xproj  = (float*)carve((size_t)MM * NG * sizeof(float));
  (void)ws_size; (void)in_sizes; (void)n_in; (void)out_size;

  (void)hipFuncSetAttribute((const void*)k_rec,
                            hipFuncAttributeMaxDynamicSharedMemorySize,
                            (int)LDS_TOTAL_B);

  hipLaunchKernelGGL(k_init, dim3(1), dim3(32), 0, stream, ctr);

  hipLaunchKernelGGL(k_pack_B, dim3((NG / 16) * (EE / 32) * 32 / 256), dim3(256),
                     0, stream, Wih0, Bih0, NG, EE);
  hipLaunchKernelGGL(k_pack_B, dim3((NG / 16) * (HH / 32) * 32 / 256), dim3(256),
                     0, stream, Whh0, Bhh0, NG, HH);
  hipLaunchKernelGGL(k_pack_B, dim3((NG / 16) * (HH / 32) * 32 / 256), dim3(256),
                     0, stream, Wih1, Bih1, NG, HH);
  hipLaunchKernelGGL(k_pack_B, dim3((NG / 16) * (HH / 32) * 32 / 256), dim3(256),
                     0, stream, Whh1, Bhh1, NG, HH);

  // Layer 0
  hipLaunchKernelGGL(k_embed_pack_A, dim3((MM / 16) * (EE / 32) * 32 / 256),
                     dim3(256), 0, stream, tok, emb, A0, MM, EE);
  hipLaunchKernelGGL((k_gemm<EE / 32>), dim3(MM / 128, NG / 128), dim3(256), 0,
                     stream, A0, Bih0, xproj, NG);
  hipLaunchKernelGGL(k_rec, dim3(16), dim3(256), LDS_TOTAL_B, stream, xproj,
                     Bhh0, h0, c0, hfr, y0, (float*)nullptr, o_hn, o_cn, ctr,
                     TT);

  // Layer 1
  hipLaunchKernelGGL(k_pack_A, dim3((MM / 16) * (HH / 32) * 32 / 256), dim3(256),
                     0, stream, y0, A1, MM, HH);
  hipLaunchKernelGGL((k_gemm<HH / 32>), dim3(MM / 128, NG / 128), dim3(256), 0,
                     stream, A1, Bih1, xproj, NG);
  hipLaunchKernelGGL(k_rec, dim3(16), dim3(256), LDS_TOTAL_B, stream, xproj,
                     Bhh1, h0 + (size_t)BB * HH, c0 + (size_t)BB * HH, hfr,
                     (float*)nullptr, pooled, o_hn + (size_t)BB * HH,
                     o_cn + (size_t)BB * HH, ctr + 1, TT);

  hipLaunchKernelGGL(k_decode, dim3(BB), dim3(256), 0, stream, pooled, decW,
                     decB, o_decoded, TT);
}
